// TreeEncoder_50405736186282
// MI455X (gfx1250) — matrix operand order repaired and checked by reading, not compiled, not run on hardware
//
#include <hip/hip_runtime.h>
#include <hip/hip_bf16.h>

#define DEVI __device__ __forceinline__

typedef __attribute__((ext_vector_type(16))) __bf16 v16bf;
typedef __attribute__((ext_vector_type(8)))  float  v8f;

constexpr int Bc    = 512;           // trees
constexpr int Nc    = 85;            // nodes per tree
constexpr int Hc    = 512;           // hidden
constexpr int Vc    = 1000;          // vocab
constexpr int NB    = 8;             // MAX_NB
constexpr int Tc    = 6;             // message passing steps
constexpr int Mtc   = 2 * (Nc - 1);  // 168 messages / tree
constexpr int Mc    = Bc * Mtc;      // 86016 total messages
constexpr int DUMI  = Mc + 1;        // scratch row
constexpr int EMAXc = Bc * 64;       // 32768 widest step

// padded LDS strides (elements) to stagger banks: 520*2B = 1040B rows
constexpr int S1 = 520;              // bf16 row stride
constexpr int SF = 520;              // f32  row stride

constexpr size_t OFF_A1   = 0;                         // [128][520] bf16
constexpr size_t OFF_SUMH = OFF_A1   + 128 * S1 * 2;   // [16][520] f32
constexpr size_t OFF_A0   = OFF_SUMH + 16 * SF * 4;    // [16][520] bf16 (sum_h)
constexpr size_t OFF_A2   = OFF_A0   + 16 * S1 * 2;    // [16][520] bf16 (sum_gated)
constexpr size_t OFF_IDX  = OFF_A2   + 16 * S1 * 2;
constexpr size_t SMEMSZ   = OFF_IDX  + (16 + 16 + 128) * 4;  // ~200 KB

DEVI const float* hsrc(const float* hm, const float* he, int r) {
  return (r < Mc) ? (hm + (size_t)r * Hc) : (he + (size_t)(r - Mc) * Hc);
}
DEVI float* hdst(float* hm, float* he, int r) {
  return (r < Mc) ? (hm + (size_t)r * Hc) : (he + (size_t)(r - Mc) * Hc);
}
DEVI float sigm(float x) {
  return __builtin_amdgcn_rcpf(1.0f + __expf(-x));     // v_exp + v_rcp only
}
DEVI float tanh_fast(float x) {
  float e = __expf(2.0f * x);                          // saturates correctly
  return 1.0f - 2.0f * __builtin_amdgcn_rcpf(e + 1.0f);
}

// ---------------------------------------------------------------------------
// Zero the PAD + DUMMY rows (workspace is poisoned 0xAA by the harness).
// ---------------------------------------------------------------------------
__global__ void init_ext_kernel(float* __restrict__ hext) {
  int i = threadIdx.x + blockIdx.x * blockDim.x;
  if (i < 2 * Hc) hext[i] = 0.0f;
}

// ---------------------------------------------------------------------------
// Transpose + convert recurrent weight blocks to bf16, N-major (WT[n*512+k])
// so a WMMA B-fragment is 16 contiguous halves per lane. Wz/Wh use the bottom
// (sum_h / sum_gated) half of the [1024,512] matrices.
// ---------------------------------------------------------------------------
__global__ void prep_w_kernel(const float* __restrict__ Wz,
                              const float* __restrict__ Ur,
                              const float* __restrict__ Wh,
                              __bf16* __restrict__ WzT,
                              __bf16* __restrict__ UrT,
                              __bf16* __restrict__ WhT) {
  int i = threadIdx.x + blockIdx.x * 256;  // i = n*512 + k
  int n = i >> 9, k = i & 511;
  WzT[i] = (__bf16)Wz[(size_t)(Hc + k) * Hc + n];
  UrT[i] = (__bf16)Ur[(size_t)k * Hc + n];
  WhT[i] = (__bf16)Wh[(size_t)(Hc + k) * Hc + n];
}

// ---------------------------------------------------------------------------
// Per-vocab x-side projections (exact f32):
//   pz[v] = emb[v] @ Wz_top + Wz_b ;  pr[v] = emb[v] @ Wr ;
//   ph[v] = emb[v] @ Wh_top + Wh_b
// ---------------------------------------------------------------------------
__global__ void __launch_bounds__(256)
prep_vocab_kernel(const float* __restrict__ emb, const float* __restrict__ Wz,
                  const float* __restrict__ Wr, const float* __restrict__ Wh,
                  const float* __restrict__ Wzb, const float* __restrict__ Whb,
                  float* __restrict__ pz, float* __restrict__ pr,
                  float* __restrict__ ph) {
  __shared__ float e[Hc];
  const int v = blockIdx.x, tid = threadIdx.x;
  e[tid]       = emb[(size_t)v * Hc + tid];
  e[tid + 256] = emb[(size_t)v * Hc + tid + 256];
  __syncthreads();
  float az0 = Wzb[tid], az1 = Wzb[tid + 256];
  float ar0 = 0.0f,     ar1 = 0.0f;
  float ah0 = Whb[tid], ah1 = Whb[tid + 256];
  for (int k = 0; k < Hc; ++k) {
    const float ev = e[k];
    const float* wz = Wz + (size_t)k * Hc;   // top half rows 0..511
    const float* wr = Wr + (size_t)k * Hc;
    const float* wh = Wh + (size_t)k * Hc;
    az0 += ev * wz[tid]; az1 += ev * wz[tid + 256];
    ar0 += ev * wr[tid]; ar1 += ev * wr[tid + 256];
    ah0 += ev * wh[tid]; ah1 += ev * wh[tid + 256];
  }
  pz[(size_t)v * Hc + tid] = az0; pz[(size_t)v * Hc + tid + 256] = az1;
  pr[(size_t)v * Hc + tid] = ar0; pr[(size_t)v * Hc + tid + 256] = ar1;
  ph[(size_t)v * Hc + tid] = ah0; ph[(size_t)v * Hc + tid + 256] = ah1;
}

// ---------------------------------------------------------------------------
// One GRU scan step. Block = 16 message rows, 8 waves, ~200KB dynamic LDS.
// Wave w owns N-tiles {4w..4w+3} for ALL matmuls; z / sum_gated partials
// stay in (statically indexed) registers between phases.
// ---------------------------------------------------------------------------
__global__ void __launch_bounds__(256)
gru_step_kernel(const int* __restrict__ sid, const int* __restrict__ msg_wid,
                const int* __restrict__ msg_nei, const float* __restrict__ pz,
                const float* __restrict__ pr, const float* __restrict__ ph,
                const float* __restrict__ urb, const __bf16* __restrict__ WzT,
                const __bf16* __restrict__ UrT, const __bf16* __restrict__ WhT,
                float* __restrict__ hmain, float* __restrict__ hext) {
  extern __shared__ char smem[];
  __bf16* A1    = (__bf16*)(smem + OFF_A1);
  float*  sumh  = (float*)(smem + OFF_SUMH);
  __bf16* A0    = (__bf16*)(smem + OFF_A0);
  __bf16* A2    = (__bf16*)(smem + OFF_A2);
  int*    s_idx = (int*)(smem + OFF_IDX);
  int*    s_wid = s_idx + 16;
  int*    s_nei = s_wid + 16;

  const int tid  = threadIdx.x;
  const int base = blockIdx.x * 16;

  if (tid < 16) s_idx[tid] = sid[base + tid];
  __syncthreads();
  if (s_idx[0] == DUMI) return;  // DUMMY entries form a contiguous tail
  if (tid < 16)  s_wid[tid] = msg_wid[s_idx[tid]];
  if (tid < 128) s_nei[tid] = msg_nei[(size_t)s_idx[tid >> 3] * NB + (tid & 7)];
  __syncthreads();

  // ---- phase 1: gather h_nei -> LDS bf16, sum_h (f32 + bf16) ----
  {
    const int c = tid * 2;
    for (int m = 0; m < 16; ++m) {
      float a0 = 0.0f, a1 = 0.0f;
      #pragma unroll
      for (int k = 0; k < NB; ++k) {
        const float* p = hsrc(hmain, hext, s_nei[m * NB + k]);
        float2 v = *(const float2*)(p + c);
        A1[(k * 16 + m) * S1 + c]     = (__bf16)v.x;
        A1[(k * 16 + m) * S1 + c + 1] = (__bf16)v.y;
        a0 += v.x; a1 += v.y;
      }
      sumh[m * SF + c] = a0;          sumh[m * SF + c + 1] = a1;
      A0[m * S1 + c]   = (__bf16)a0;  A0[m * S1 + c + 1]   = (__bf16)a1;
    }
  }
  __syncthreads();

  const int w = tid >> 5, lane = tid & 31;
  const int lanelo = lane & 15, hi = lane >> 4, klo = hi * 16;

  // ---- phase 2a: z = sigmoid(pz + sum_h @ WzT) ----
  float zreg[4][8];
  #pragma unroll
  for (int ctl = 0; ctl < 4; ++ctl) {
    const int n = (w * 4 + ctl) * 16 + lanelo;
    v8f acc = {0.f, 0.f, 0.f, 0.f, 0.f, 0.f, 0.f, 0.f};
    #pragma unroll 4
    for (int kk = 0; kk < 16; ++kk) {
      const int kb = kk * 32;
      const __bf16* ap = A0 + lanelo * S1 + kb + klo;
      v16bf a;
      #pragma unroll
      for (int i = 0; i < 8; ++i) { a[i] = ap[i]; a[i + 8] = ap[i + 8]; }
      v16bf b = *(const v16bf*)(WzT + (size_t)n * Hc + kb + hi * 16);
      acc = __builtin_amdgcn_wmma_f32_16x16x32_bf16(false, a, false, b, (short)0, acc, false, false);
    }
    #pragma unroll
    for (int v = 0; v < 8; ++v) {
      const int m = v + 8 * hi;
      zreg[ctl][v] = sigm(acc[v] + pz[(size_t)s_wid[m] * Hc + n]);
    }
  }

  // ---- phase 2b: r_k = sigmoid(pr + h_nei_k @ UrT + Ur_b); sg = sum r*h ----
  float sg[4][8], xr[4][8], urbl[4];
  #pragma unroll
  for (int ctl = 0; ctl < 4; ++ctl) {
    const int n = (w * 4 + ctl) * 16 + lanelo;
    urbl[ctl] = urb[n];
    #pragma unroll
    for (int v = 0; v < 8; ++v) {
      xr[ctl][v] = pr[(size_t)s_wid[v + 8 * hi] * Hc + n];
      sg[ctl][v] = 0.0f;
    }
  }
  #pragma unroll 1
  for (int k = 0; k < NB; ++k) {
    #pragma unroll
    for (int ctl = 0; ctl < 4; ++ctl) {
      const int n = (w * 4 + ctl) * 16 + lanelo;
      v8f acc = {0.f, 0.f, 0.f, 0.f, 0.f, 0.f, 0.f, 0.f};
      #pragma unroll 4
      for (int kk = 0; kk < 16; ++kk) {
        const int kb = kk * 32;
        const __bf16* ap = A1 + (k * 16 + lanelo) * S1 + kb + klo;
        v16bf a;
        #pragma unroll
        for (int i = 0; i < 8; ++i) { a[i] = ap[i]; a[i + 8] = ap[i + 8]; }
        v16bf b = *(const v16bf*)(UrT + (size_t)n * Hc + kb + hi * 16);
        acc = __builtin_amdgcn_wmma_f32_16x16x32_bf16(false, a, false, b, (short)0, acc, false, false);
      }
      #pragma unroll
      for (int v = 0; v < 8; ++v) {
        const int m = v + 8 * hi;
        float rv = sigm(acc[v] + xr[ctl][v] + urbl[ctl]);
        float hn = (float)A1[(k * 16 + m) * S1 + n];
        sg[ctl][v] += rv * hn;
      }
    }
  }
  #pragma unroll
  for (int ctl = 0; ctl < 4; ++ctl) {
    const int n = (w * 4 + ctl) * 16 + lanelo;
    #pragma unroll
    for (int v = 0; v < 8; ++v) A2[(v + 8 * hi) * S1 + n] = (__bf16)sg[ctl][v];
  }
  __syncthreads();

  // ---- phase 3: pre_h = tanh(ph + sg @ WhT); new_h = (1-z)*sum_h + z*pre ----
  #pragma unroll
  for (int ctl = 0; ctl < 4; ++ctl) {
    const int n = (w * 4 + ctl) * 16 + lanelo;
    v8f acc = {0.f, 0.f, 0.f, 0.f, 0.f, 0.f, 0.f, 0.f};
    #pragma unroll 4
    for (int kk = 0; kk < 16; ++kk) {
      const int kb = kk * 32;
      const __bf16* ap = A2 + lanelo * S1 + kb + klo;
      v16bf a;
      #pragma unroll
      for (int i = 0; i < 8; ++i) { a[i] = ap[i]; a[i + 8] = ap[i + 8]; }
      v16bf b = *(const v16bf*)(WhT + (size_t)n * Hc + kb + hi * 16);
      acc = __builtin_amdgcn_wmma_f32_16x16x32_bf16(false, a, false, b, (short)0, acc, false, false);
    }
    #pragma unroll
    for (int v = 0; v < 8; ++v) {
      const int m = v + 8 * hi;
      float pre = tanh_fast(acc[v] + ph[(size_t)s_wid[m] * Hc + n]);
      float zz  = zreg[ctl][v];
      float out = (1.0f - zz) * sumh[m * SF + n] + zz * pre;
      hdst(hmain, hext, s_idx[m])[n] = out;
    }
  }
}

// ---------------------------------------------------------------------------
// Root aggregation: relu(cat(emb[root_wid], sum h[root_nei]) @ W_w + W_b)
// ---------------------------------------------------------------------------
__global__ void __launch_bounds__(256)
root_kernel(const int* __restrict__ root_wid, const int* __restrict__ root_nei,
            const float* __restrict__ emb, const float* __restrict__ Ww,
            const float* __restrict__ Wb, const float* __restrict__ hmain,
            const float* __restrict__ hext, float* __restrict__ out_root) {
  __shared__ float cat[2 * Hc];
  const int b = blockIdx.x, tid = threadIdx.x;
  const int wid = root_wid[b];
  for (int i = tid; i < Hc; i += 256) {
    cat[i] = emb[(size_t)wid * Hc + i];
    float s = 0.0f;
    #pragma unroll
    for (int k = 0; k < NB; ++k) {
      const float* p = hsrc(hmain, hext, root_nei[b * NB + k]);
      s += p[i];
    }
    cat[Hc + i] = s;
  }
  __syncthreads();
  float a0 = Wb[tid], a1 = Wb[tid + 256];
  for (int k = 0; k < 2 * Hc; ++k) {
    const float cv = cat[k];
    a0 += cv * Ww[(size_t)k * Hc + tid];
    a1 += cv * Ww[(size_t)k * Hc + tid + 256];
  }
  out_root[(size_t)b * Hc + tid]       = a0 > 0.0f ? a0 : 0.0f;
  out_root[(size_t)b * Hc + tid + 256] = a1 > 0.0f ? a1 : 0.0f;
}

// ---------------------------------------------------------------------------
extern "C" void kernel_launch(void* const* d_in, const int* in_sizes, int n_in,
                              void* d_out, int out_size, void* d_ws,
                              size_t ws_size, hipStream_t stream) {
  (void)in_sizes; (void)n_in; (void)out_size; (void)ws_size;
  const int*   msg_wid  = (const int*)d_in[0];
  const int*   msg_nei  = (const int*)d_in[1];
  const int*   step_ids = (const int*)d_in[2];
  const int*   root_wid = (const int*)d_in[3];
  const int*   root_nei = (const int*)d_in[4];
  const float* emb      = (const float*)d_in[5];
  const float* Wz_w     = (const float*)d_in[6];
  const float* Wz_b     = (const float*)d_in[7];
  const float* Wr_w     = (const float*)d_in[8];
  const float* Ur_w     = (const float*)d_in[9];
  const float* Ur_b     = (const float*)d_in[10];
  const float* Wh_w     = (const float*)d_in[11];
  const float* Wh_b     = (const float*)d_in[12];
  const float* W_w      = (const float*)d_in[13];
  const float* W_b      = (const float*)d_in[14];

  char* ws = (char*)d_ws;
  float*  hext = (float*)ws;  ws += 4096;                 // PAD + DUMMY rows
  float*  pz   = (float*)ws;  ws += (size_t)Vc * Hc * 4;
  float*  pr   = (float*)ws;  ws += (size_t)Vc * Hc * 4;
  float*  ph   = (float*)ws;  ws += (size_t)Vc * Hc * 4;
  __bf16* WzT  = (__bf16*)ws; ws += (size_t)Hc * Hc * 2;
  __bf16* UrT  = (__bf16*)ws; ws += (size_t)Hc * Hc * 2;
  __bf16* WhT  = (__bf16*)ws;

  float* hmain    = (float*)d_out;                  // h rows [0, M) == output 1
  float* out_root = (float*)d_out + (size_t)Mc * Hc;

  init_ext_kernel<<<4, 256, 0, stream>>>(hext);
  prep_w_kernel<<<(Hc * Hc) / 256, 256, 0, stream>>>(Wz_w, Ur_w, Wh_w, WzT, UrT, WhT);
  prep_vocab_kernel<<<Vc, 256, 0, stream>>>(emb, Wz_w, Wr_w, Wh_w, Wz_b, Wh_b, pz, pr, ph);

  for (int t = 0; t < Tc; ++t) {
    gru_step_kernel<<<EMAXc / 16, 256, SMEMSZ, stream>>>(
        step_ids + (size_t)t * EMAXc, msg_wid, msg_nei, pz, pr, ph, Ur_b,
        WzT, UrT, WhT, hmain, hext);
  }
  root_kernel<<<Bc, 256, 0, stream>>>(root_wid, root_nei, emb, W_w, W_b,
                                      hmain, hext, out_root);
}
